// Linear4bit_584115553009
// MI455X (gfx1250) — compile-verified, matrix operand order checked by reading, not run on hardware
//
#include <hip/hip_runtime.h>
#include <hip/hip_bf16.h>
#include <stdint.h>

typedef __attribute__((ext_vector_type(16))) _Float16 v16h;
typedef __attribute__((ext_vector_type(8)))  _Float16 v8h;
typedef __attribute__((ext_vector_type(2)))  _Float16 h2;
typedef __attribute__((ext_vector_type(8)))  float    v8f;

#define KDIM 4096
#define OUTF 11008
#define BM   128
#define BN   128
#define BK   64
#define LDSP 72   // padded f16 row stride (64 + 8) to dodge bank conflicts

__device__ __constant__ float NF4_TAB[16] = {
    -1.0f, -0.6961928009986877f, -0.5250730514526367f, -0.39491748809814453f,
    -0.28444138169288635f, -0.18477343022823334f, -0.09105003625154495f, 0.0f,
    0.07958029955625534f, 0.16093020141124725f, 0.24611230194568634f,
    0.33791524171829224f, 0.44070982933044434f, 0.5626170039176941f,
    0.8072066307067871f, 1.0f };

// On gfx1250 the flat LDS aperture keeps the LDS byte offset in the low 32
// address bits, so truncating the generic pointer gives the DS address.
__device__ __forceinline__ uint32_t lds_off(const void* p) {
    return (uint32_t)(uintptr_t)p;
}

// CDNA5 async global->LDS copy (tracked by ASYNCcnt).
__device__ __forceinline__ void async_copy_b128(const void* gaddr, uint32_t ldsaddr) {
    asm volatile("global_load_async_to_lds_b128 %0, %1, off"
                 :: "v"(ldsaddr), "v"(gaddr) : "memory");
}

__global__ __launch_bounds__(256)
void nf4_linear_wmma(const _Float16* __restrict__ x,
                     const int*      __restrict__ wpacked,
                     const float*    __restrict__ wabsmax,
                     const _Float16* __restrict__ bias,
                     float*          __restrict__ out)
{
    __shared__ _Float16 Atile[BM][LDSP];   // x tile: 128 tokens x 64 K
    __shared__ _Float16 Btile[BN][LDSP];   // dequantized W tile: 128 outs x 64 K
    __shared__ uint32_t lut[256];          // byte -> packed {hi,lo} NF4 f16 pair

    const int tid  = threadIdx.x;
    const int lane = tid & 31;
    const int wave = tid >> 5;
    const int wm   = wave >> 1;        // 0..3: 32-row band
    const int wn   = wave & 1;         // 0..1: 64-col band
    const int lrow = lane & 15;
    const int sel  = lane >> 4;        // ISA f16 fragment lane-half select

    const int n0 = blockIdx.x * BN;    // out-feature base
    const int m0 = blockIdx.y * BM;    // token base

    // --- build byte->f16-pair LUT once (high nibble = even/first element) ---
    {
        const int b = tid;                       // blockDim == 256
        h2 p;
        p[0] = (_Float16)NF4_TAB[(b >> 4) & 0xF];
        p[1] = (_Float16)NF4_TAB[b & 0xF];
        lut[b] = *(const uint32_t*)&p;
    }
    __syncthreads();

    v8f acc[2][4] = {};

    // B-tile producer role: each thread owns half a row (16 packed bytes)
    const int n_local = tid >> 1;      // 0..127
    const int half    = tid & 1;       // which 16-byte half of the 32 bytes
    const int*   wrow = wpacked + (size_t)(n0 + n_local) * (KDIM / 2);
    const float* warow = wabsmax + (size_t)(n0 + n_local) * (KDIM / BK);

    for (int k0 = 0; k0 < KDIM; k0 += BK) {
        // ---- async-copy A tile: 128 x 64 f16 = 1024 x 16B chunks ----
        #pragma unroll
        for (int i = 0; i < 4; ++i) {
            const int c   = tid + i * 256;       // chunk id
            const int row = c >> 3;              // 8 chunks per row
            const int c8  = c & 7;               // chunk within row (8 f16 each)
            const _Float16* g = x + (size_t)(m0 + row) * KDIM + k0 + c8 * 8;
            async_copy_b128(g, lds_off(&Atile[row][c8 * 8]));
        }

        // ---- dequantize W tile: LUT + per-block absmax scale ----
        {
            const float s = warow[k0 >> 6];
            h2 s2; s2[0] = (_Float16)s; s2[1] = (_Float16)s;
            const int4* wp16 = (const int4*)(wrow + (k0 >> 1) + half * 16);
            if (k0 + BK < KDIM)   // speculative prefetch of next weight chunk
                __builtin_prefetch(wrow + ((k0 + BK) >> 1) + half * 16, 0, 0);
            #pragma unroll
            for (int q = 0; q < 4; ++q) {
                const int4 w4 = wp16[q];
                const int vals[4] = { w4.x, w4.y, w4.z, w4.w };
                #pragma unroll
                for (int j = 0; j < 4; ++j) {
                    const uint32_t pk = lut[(uint32_t)vals[j] & 0xFFu];
                    h2 v = *(const h2*)&pk;
                    v = v * s2;                            // v_pk_mul_f16
                    const int kloc = half * 32 + q * 8 + j * 2;
                    *(h2*)&Btile[n_local][kloc] = v;
                }
            }
        }

        asm volatile("s_wait_asynccnt 0" ::: "memory");
        __syncthreads();

        // ---- matrix core: 2 k-slices of 32, 8 WMMAs each ----
        #pragma unroll
        for (int kk = 0; kk < BK; kk += 32) {
            v16h afrag[2], bfrag[4];
            #pragma unroll
            for (int mt = 0; mt < 2; ++mt) {
                const _Float16* ap = &Atile[wm * 32 + mt * 16 + lrow][kk + sel * 8];
                const v8h alo = *(const v8h*)ap;
                const v8h ahi = *(const v8h*)(ap + 16);
                #pragma unroll
                for (int i = 0; i < 8; ++i) { afrag[mt][i] = alo[i]; afrag[mt][8 + i] = ahi[i]; }
            }
            #pragma unroll
            for (int nt = 0; nt < 4; ++nt) {
                const _Float16* bp = &Btile[wn * 64 + nt * 16 + lrow][kk + sel * 8];
                const v8h blo = *(const v8h*)bp;
                const v8h bhi = *(const v8h*)(bp + 16);
                #pragma unroll
                for (int i = 0; i < 8; ++i) { bfrag[nt][i] = blo[i]; bfrag[nt][8 + i] = bhi[i]; }
            }
            #pragma unroll
            for (int mt = 0; mt < 2; ++mt)
                #pragma unroll
                for (int nt = 0; nt < 4; ++nt)
                    acc[mt][nt] = __builtin_amdgcn_wmma_f32_16x16x32_f16(
                        false, afrag[mt], false, bfrag[nt],
                        (short)0, acc[mt][nt], false, false);
        }
        __syncthreads();
    }

    // ---- epilogue: bias add + store (C/D layout: VGPR r -> M=r / r+8) ----
    #pragma unroll
    for (int nt = 0; nt < 4; ++nt) {
        const int col = n0 + wn * 64 + nt * 16 + lrow;
        const float bv = (float)bias[col];
        #pragma unroll
        for (int mt = 0; mt < 2; ++mt) {
            const int rbase = m0 + wm * 32 + mt * 16 + sel * 8;
            #pragma unroll
            for (int r = 0; r < 8; ++r)
                out[(size_t)(rbase + r) * OUTF + col] = acc[mt][nt][r] + bv;
        }
    }
}

extern "C" void kernel_launch(void* const* d_in, const int* in_sizes, int n_in,
                              void* d_out, int out_size, void* d_ws, size_t ws_size,
                              hipStream_t stream) {
    const _Float16* x    = (const _Float16*)d_in[0];
    const int*      wp   = (const int*)d_in[1];
    const float*    wa   = (const float*)d_in[2];
    const _Float16* bias = (const _Float16*)d_in[3];
    float* out = (float*)d_out;   // non-bf16 output -> float* per harness contract

    const int ntok = in_sizes[0] / KDIM;          // 4*1024 = 4096
    dim3 grid(OUTF / BN, ntok / BM);              // (86, 32)
    nf4_linear_wmma<<<grid, 256, 0, stream>>>(x, wp, wa, bias, out);

    (void)n_in; (void)out_size; (void)d_ws; (void)ws_size;
}
